// EncoderBlock_48163763258146
// MI455X (gfx1250) — matrix operand order repaired and checked by reading, not compiled, not run on hardware
//
#include <hip/hip_runtime.h>
#include <cstdint>
#include <cstddef>

// ---------------------------------------------------------------------------
// Problem constants (from reference)
// ---------------------------------------------------------------------------
static constexpr int Bb   = 4;
static constexpr int Hh   = 64;
static constexpr int Ww   = 64;
static constexpr int Cc   = 768;
static constexpr int NH   = 12;
static constexpr int HD   = 64;
static constexpr int WS   = 14;
static constexpr int MLP  = 3072;
static constexpr int NWH  = 5;              // ceil(64/14)
static constexpr int NWW  = 5;
static constexpr int BW   = Bb * NWH * NWW; // 100 windows
static constexpr int NT   = WS * WS;        // 196 tokens per window
static constexpr int ROWS = BW * NT;        // 19600  (multiple of 16)
static constexpr int TOK  = Bb * Hh * Ww;   // 16384
static constexpr float SCALE = 0.125f;      // 64^-0.5

typedef __bf16 bf16_t;
typedef __attribute__((ext_vector_type(16))) __bf16 v16bf;
typedef __attribute__((ext_vector_type(8)))  float  v8f;

union Frag { uint4 u[2]; v16bf v; };

// D = A(16x32 bf16) * B(32x16 bf16) + C(16x16 f32)
__device__ __forceinline__ v8f wmma_bf16(v16bf a, v16bf b, v8f c) {
  return __builtin_amdgcn_wmma_f32_16x16x32_bf16(false, a, false, b, (short)0, c,
                                                 false, false);
}

// A-matrix 16x32 bf16 layout (ISA 7.12.2): lanes 0-15 row M=lane, K = {0..7,16..23};
// lanes 16-31 same rows, K = {8..15,24..31}.  rowk = &A[row][k0].
__device__ __forceinline__ v16bf load_a_frag(const bf16_t* rowk, int lane) {
  const bf16_t* p = rowk + ((lane & 16) ? 8 : 0);
  Frag f;
  f.u[0] = *(const uint4*)(p);
  f.u[1] = *(const uint4*)(p + 16);
  return f.v;
}

// B-matrix 32x16 bf16: lane%16 = column, lanes 0-15 hold K = {0..7,16..23},
// lanes 16-31 hold K = {8..15,24..31} (same K order as A).
// rowk = &Bt[col][k0]  (Bt is B^T, row-major [N][K]).
__device__ __forceinline__ v16bf load_b_frag(const bf16_t* rowk, int lane) {
  const bf16_t* p = rowk + ((lane & 16) ? 8 : 0);
  Frag f;
  f.u[0] = *(const uint4*)(p);
  f.u[1] = *(const uint4*)(p + 16);
  return f.v;
}

// ---------------------------------------------------------------------------
// gfx1250 async global->LDS DMA (ASYNCcnt-tracked).  ldsoff = 32-bit LDS byte
// address (low 32 bits of a generic LDS pointer), voff = per-lane byte offset,
// base = 64-bit SGPR base (GVS mode: mem = SGPR + VGPR + ioffset).
// ---------------------------------------------------------------------------
__device__ __forceinline__ void async_b128_to_lds(unsigned ldsoff, unsigned voff,
                                                  unsigned long long base) {
  asm volatile("global_load_async_to_lds_b128 %0, %1, %2"
               :: "v"(ldsoff), "v"(voff), "s"(base)
               : "memory");
}
__device__ __forceinline__ void wait_asynccnt0() {
  asm volatile("s_wait_asynccnt 0x0" ::: "memory");
}

// ---------------------------------------------------------------------------
// Weight prep: W[K,N] f32 -> Wt[N,K] bf16
// ---------------------------------------------------------------------------
__global__ __launch_bounds__(256) void transpose_bf16_kernel(
    const float* __restrict__ in, bf16_t* __restrict__ out, int K, int N) {
  size_t i = (size_t)blockIdx.x * 256 + threadIdx.x;
  if (i >= (size_t)K * N) return;
  int k = (int)(i / N), n = (int)(i % N);
  out[(size_t)n * K + k] = (bf16_t)in[i];
}

// ---------------------------------------------------------------------------
// LN1 + window partition -> xw bf16 [ROWS, C].  Padded tokens -> zero rows.
// ---------------------------------------------------------------------------
__global__ __launch_bounds__(256) void ln1_partition_kernel(
    const float* __restrict__ x, const float* __restrict__ g,
    const float* __restrict__ b, bf16_t* __restrict__ out) {
  int row = blockIdx.x;                       // w_idx*196 + n
  int w_idx = row / NT, n = row % NT;
  int bb = w_idx / (NWH * NWW);
  int wh = (w_idx % (NWH * NWW)) / NWW;
  int ww = w_idx % NWW;
  int i = n / WS, j = n % WS;
  int h = wh * WS + i, w = ww * WS + j;
  bool valid = (h < Hh) && (w < Ww);

  __shared__ float2 red[8];
  int tid = threadIdx.x, lane = tid & 31, wave = tid >> 5;
  const float* xr = x + (((size_t)bb * Hh + h) * Ww + w) * Cc;
  float vals[3] = {0.f, 0.f, 0.f};
  float s = 0.f, s2 = 0.f;
  if (valid) {
    for (int t = 0; t < 3; ++t) {
      float v = xr[tid + 256 * t];
      vals[t] = v; s += v; s2 += v * v;
    }
  }
  for (int o = 16; o > 0; o >>= 1) { s += __shfl_xor(s, o, 32); s2 += __shfl_xor(s2, o, 32); }
  if (lane == 0) red[wave] = make_float2(s, s2);
  __syncthreads();
  if (tid == 0) {
    float a = 0.f, c = 0.f;
    for (int k = 0; k < 8; ++k) { a += red[k].x; c += red[k].y; }
    red[0] = make_float2(a, c);
  }
  __syncthreads();
  float mean = red[0].x * (1.0f / Cc);
  float var  = red[0].y * (1.0f / Cc) - mean * mean;
  float inv  = rsqrtf(var + 1e-6f);
  bf16_t* orow = out + (size_t)row * Cc;
  for (int t = 0; t < 3; ++t) {
    int c = tid + 256 * t;
    float v = valid ? ((vals[t] - mean) * inv * g[c] + b[c]) : 0.f;
    orow[c] = (bf16_t)v;
  }
}

// ---------------------------------------------------------------------------
// LN2 on x1 [TOK, C] f32 -> bf16
// ---------------------------------------------------------------------------
__global__ __launch_bounds__(256) void ln2_kernel(
    const float* __restrict__ x1, const float* __restrict__ g,
    const float* __restrict__ b, bf16_t* __restrict__ out) {
  int row = blockIdx.x;
  __shared__ float2 red[8];
  int tid = threadIdx.x, lane = tid & 31, wave = tid >> 5;
  const float* xr = x1 + (size_t)row * Cc;
  float vals[3]; float s = 0.f, s2 = 0.f;
  for (int t = 0; t < 3; ++t) {
    float v = xr[tid + 256 * t];
    vals[t] = v; s += v; s2 += v * v;
  }
  for (int o = 16; o > 0; o >>= 1) { s += __shfl_xor(s, o, 32); s2 += __shfl_xor(s2, o, 32); }
  if (lane == 0) red[wave] = make_float2(s, s2);
  __syncthreads();
  if (tid == 0) {
    float a = 0.f, c = 0.f;
    for (int k = 0; k < 8; ++k) { a += red[k].x; c += red[k].y; }
    red[0] = make_float2(a, c);
  }
  __syncthreads();
  float mean = red[0].x * (1.0f / Cc);
  float var  = red[0].y * (1.0f / Cc) - mean * mean;
  float inv  = rsqrtf(var + 1e-6f);
  bf16_t* orow = out + (size_t)row * Cc;
  for (int t = 0; t < 3; ++t) {
    int c = tid + 256 * t;
    orow[c] = (bf16_t)((vals[t] - mean) * inv * g[c] + b[c]);
  }
}

// ---------------------------------------------------------------------------
// Generic WMMA GEMM: C[M,N] = A[M,K](bf16) * Bt[N,K]^T(bf16) + bias, + epilogue
// block = 256 threads (8 waves); wave tile = 16(M) x 64(N); block tile 128x64.
// The shared 64x32 B tile is staged in LDS via double-buffered
// global_load_async_to_lds_b128 (ASYNCcnt), overlapping DMA with WMMA.
// MODE 0: qkv split   MODE 1: proj+residual+unpartition
// MODE 2: fc1+GELU    MODE 3: fc2+residual -> d_out
// ---------------------------------------------------------------------------
template <int MODE>
__global__ __launch_bounds__(256) void gemm_bf16_kernel(
    const bf16_t* __restrict__ A, const bf16_t* __restrict__ Bt,
    const float* __restrict__ bias, int M, int N, int K,
    float* outF, bf16_t* outB, const float* __restrict__ res,
    bf16_t* qb, bf16_t* kb, bf16_t* vb) {
  __shared__ bf16_t btile[2][64 * 32];   // two 4KB K-step tiles of B^T

  int tid = threadIdx.x, lane = tid & 31, wave = tid >> 5;
  int m0 = blockIdx.x * 128 + wave * 16;
  int n0 = blockIdx.y * 64;

  int arow = m0 + (lane & 15);
  if (arow >= M) arow = M - 1;                 // clamp (stores are guarded)
  const bf16_t* aptr = A + (size_t)arow * K;

  // async B-tile staging: thread t -> row t/4 (of 64), 16B chunk t%4
  int brow = tid >> 2, bchunk = tid & 3;
  unsigned long long bbase = (unsigned long long)(uintptr_t)Bt;
  unsigned voff = (unsigned)(((size_t)(n0 + brow) * K + bchunk * 8) * 2);
  unsigned lds0 = (unsigned)(size_t)&btile[0][brow * 32 + bchunk * 8];
  unsigned lds1 = (unsigned)(size_t)&btile[1][brow * 32 + bchunk * 8];

  async_b128_to_lds(lds0, voff, bbase);        // prologue: tile 0

  v8f acc[4] = {};
  int it = 0;
  for (int k0 = 0; k0 < K; k0 += 32, ++it) {
    wait_asynccnt0();
    __syncthreads();                            // tile (it&1) ready, prev reads done
    if (k0 + 32 < K)                            // prefetch next tile into other buf
      async_b128_to_lds((it & 1) ? lds0 : lds1,
                        voff + (unsigned)((k0 + 32) * 2), bbase);
    const bf16_t* tile = btile[it & 1];
    v16bf fa = load_a_frag(aptr + k0, lane);
    for (int j = 0; j < 4; ++j) {
      const bf16_t* bp = tile + (j * 16 + (lane & 15)) * 32;
      v16bf fb = load_b_frag(bp, lane);
      acc[j] = wmma_bf16(fa, fb, acc[j]);
    }
  }

  // C/D layout: VGPR r -> row m0 + r + 8*(lane>=16), col n0 + j*16 + lane%16
  int rbase = m0 + ((lane >> 4) << 3);
  int cbase = n0 + (lane & 15);
  for (int j = 0; j < 4; ++j) {
    int col = cbase + j * 16;
    float bv = bias[col];
    for (int r = 0; r < 8; ++r) {
      int row = rbase + r;
      if (row >= M) continue;
      float v = acc[j][r] + bv;
      if constexpr (MODE == 0) {
        // row = w_idx*196+n ; col -> {q|k|v}, head, d ; head-major output
        int w_idx = row / NT, n = row % NT;
        int which = col / Cc;
        int rem = col - which * Cc;
        int head = rem >> 6, d = rem & 63;
        size_t dst = ((size_t)(w_idx * NH + head) * NT + n) * HD + d;
        bf16_t bvv = (bf16_t)v;
        if (which == 0)      qb[dst] = bvv;
        else if (which == 1) kb[dst] = bvv;
        else                 vb[dst] = bvv;
      } else if constexpr (MODE == 1) {
        // proj: unpartition + crop + residual -> x1 f32
        int w_idx = row / NT, n = row % NT;
        int bb = w_idx / (NWH * NWW);
        int wh = (w_idx % (NWH * NWW)) / NWW;
        int ww = w_idx % NWW;
        int h = wh * WS + n / WS, w = ww * WS + n % WS;
        if (h < Hh && w < Ww) {
          size_t idx = (((size_t)bb * Hh + h) * Ww + w) * Cc + col;
          outF[idx] = res[idx] + v;
        }
      } else if constexpr (MODE == 2) {
        // exact GELU (erf)
        float gelu = 0.5f * v * (1.0f + erff(v * 0.70710678118654752f));
        outB[(size_t)row * MLP + col] = (bf16_t)gelu;
      } else {
        size_t idx = (size_t)row * Cc + col;
        outF[idx] = res[idx] + v;
      }
    }
  }
}

// ---------------------------------------------------------------------------
// Attention: one block per (window, head) = 1200 blocks, 256 threads.
// Everything (q,k,vT, rel tables, full 208x208 f32 score matrix) lives in LDS
// (~284 KB of the 320 KB WGP LDS).  QK^T and P*V use WMMA bf16; q/k tiles are
// staged with async global->LDS DMA.
// ---------------------------------------------------------------------------
static constexpr int NP   = 208;   // 196 padded to 13*16 (rows/cols of S)
static constexpr int SPAD = 224;   // K padding for the P*V GEMM (7 * 32)
// LDS layout (bytes, all 16-aligned)
static constexpr int OFF_Q    = 0;                       // 208*64*2  = 26624
static constexpr int OFF_K    = OFF_Q + 26624;           // 26624
static constexpr int OFF_VT   = OFF_K + 26624;           // 64*224*2  = 28672
static constexpr int OFF_RH   = OFF_VT + 28672;          // 27*64*4   = 6912
static constexpr int OFF_RW   = OFF_RH + 6912;           // 6912
static constexpr int OFF_RELH = OFF_RW + 6912;           // 196*14*4  = 10976
static constexpr int OFF_RELW = OFF_RELH + 10976;        // 10976
static constexpr int OFF_S    = OFF_RELW + 10976;        // 208*208*4 = 173056
static constexpr int ATT_SMEM = OFF_S + NP * NP * 4;     // 290752 bytes

__global__ __launch_bounds__(256) void attn_kernel(
    const bf16_t* __restrict__ Q, const bf16_t* __restrict__ Kt,
    const bf16_t* __restrict__ V, const float* __restrict__ rel_h,
    const float* __restrict__ rel_w, bf16_t* __restrict__ Obuf) {
  extern __shared__ char smem[];
  bf16_t* q_s  = (bf16_t*)(smem + OFF_Q);    // [208][64]
  bf16_t* k_s  = (bf16_t*)(smem + OFF_K);    // [208][64]
  bf16_t* vT   = (bf16_t*)(smem + OFF_VT);   // [64][224]
  float*  Rh_s = (float*)(smem + OFF_RH);    // [27][64]
  float*  Rw_s = (float*)(smem + OFF_RW);    // [27][64]
  float*  relh = (float*)(smem + OFF_RELH);  // [196][14]
  float*  relw = (float*)(smem + OFF_RELW);  // [196][14]
  float*  S    = (float*)(smem + OFF_S);     // [208][208] f32
  bf16_t* P    = (bf16_t*)(smem + OFF_S);    // aliased: row stride 416 bf16

  int bh = blockIdx.x;
  size_t base = (size_t)bh * (NT * HD);
  int tid = threadIdx.x, lane = tid & 31, wave = tid >> 5;

  // ---- stage q, k via async DMA; v transposed + rel tables via VALU -------
  {
    unsigned long long qg = (unsigned long long)(uintptr_t)(Q + base);
    unsigned long long kg = (unsigned long long)(uintptr_t)(Kt + base);
    unsigned qld = (unsigned)(size_t)q_s;
    unsigned kld = (unsigned)(size_t)k_s;
    for (int c = tid; c < (NT * HD) / 8; c += 256) {   // 1568 x 16B chunks
      unsigned off = (unsigned)(c * 16);
      async_b128_to_lds(qld + off, off, qg);
      async_b128_to_lds(kld + off, off, kg);
    }
    // zero the padded rows 196..207 (dword stores)
    unsigned* q32 = (unsigned*)q_s;
    unsigned* k32 = (unsigned*)k_s;
    for (int i = NT * 32 + tid; i < NP * 32; i += 256) { q32[i] = 0u; k32[i] = 0u; }
    for (int i = tid; i < HD * SPAD; i += 256) {
      int d = i / SPAD, n = i % SPAD;
      bf16_t val = (bf16_t)0.f;
      if (n < NT) val = V[base + (size_t)n * HD + d];
      vT[i] = val;
    }
    for (int i = tid; i < 27 * 64; i += 256) { Rh_s[i] = rel_h[i]; Rw_s[i] = rel_w[i]; }
  }
  wait_asynccnt0();
  __syncthreads();

  // ---- decomposed relative-position projections (small, VALU) -------------
  for (int t = tid; t < NT * WS; t += 256) {
    int m = t / WS, kk = t % WS;
    int qi = m / WS, qj = m % WS;
    const bf16_t* qr = q_s + m * HD;
    const float* rh = Rh_s + (qi - kk + WS - 1) * HD;
    const float* rw = Rw_s + (qj - kk + WS - 1) * HD;
    float sh = 0.f, sw = 0.f;
    for (int c = 0; c < HD; ++c) {
      float qv = (float)qr[c];
      sh += qv * rh[c]; sw += qv * rw[c];
    }
    relh[t] = sh; relw[t] = sw;
  }
  __syncthreads();

  // ---- S = q k^T (WMMA): 13x13 tiles over 8 waves -------------------------
  for (int tile = wave; tile < 13 * 13; tile += 8) {
    int ti = tile / 13, tj = tile % 13;
    const bf16_t* ap = q_s + (ti * 16 + (lane & 15)) * HD;
    const bf16_t* bp = k_s + (tj * 16 + (lane & 15)) * HD;
    v8f acc = {};
    for (int k0 = 0; k0 < HD; k0 += 32) {
      v16bf fa = load_a_frag(ap + k0, lane);
      v16bf fb = load_b_frag(bp + k0, lane);
      acc = wmma_bf16(fa, fb, acc);
    }
    int rb = ti * 16 + ((lane >> 4) << 3);
    int cc = tj * 16 + (lane & 15);
    for (int r = 0; r < 8; ++r) S[(rb + r) * NP + cc] = acc[r];
  }
  __syncthreads();

  // ---- softmax rows with rel-pos bias; convert to bf16 in place -----------
  // (padded q-rows are zero => S rows 196..207 are exact zeros => P rows
  //  196..207 read back as bf16 zeros; cols 196..223 zeroed explicitly)
  if (tid < NT) {
    int m = tid;
    float* srow = S + m * NP;
    const float* rh = relh + m * WS;
    const float* rw = relw + m * WS;
    float mx = -3.0e38f;
    for (int n = 0; n < NT; ++n) {
      float l = srow[n] * SCALE + rh[n / WS] + rw[n % WS];
      mx = fmaxf(mx, l);
    }
    float sum = 0.f;
    for (int n = 0; n < NT; ++n) {
      float l = srow[n] * SCALE + rh[n / WS] + rw[n % WS];
      float e = __expf(l - mx);
      srow[n] = e; sum += e;
    }
    float inv = 1.0f / sum;
    // forward in-place f32 -> bf16 compaction within this row only;
    // volatile on both sides forbids any reordering of the aliased accesses.
    volatile float*  sv = srow;
    volatile bf16_t* pv = P + m * (2 * NP);   // row stride 416 bf16 elems
    for (int n = 0; n < NT; ++n) {
      float e = sv[n];
      pv[n] = (bf16_t)(e * inv);
    }
    for (int n = NT; n < SPAD; ++n) pv[n] = (bf16_t)0.f;
  }
  __syncthreads();

  // ---- O = P * v (WMMA): 13 row tiles x 4 col tiles ------------------------
  int w_idx = bh / NH, head = bh % NH;
  for (int ti = wave; ti < 13; ti += 8) {
    const bf16_t* ap = P + (ti * 16 + (lane & 15)) * (2 * NP);
    v8f acc[4] = {};
    for (int k0 = 0; k0 < SPAD; k0 += 32) {
      v16bf fa = load_a_frag(ap + k0, lane);
      for (int j = 0; j < 4; ++j) {
        const bf16_t* bp = vT + (j * 16 + (lane & 15)) * SPAD;
        v16bf fb = load_b_frag(bp + k0, lane);
        acc[j] = wmma_bf16(fa, fb, acc[j]);
      }
    }
    int rb = ti * 16 + ((lane >> 4) << 3);
    for (int j = 0; j < 4; ++j) {
      int d = j * 16 + (lane & 15);
      for (int r = 0; r < 8; ++r) {
        int m = rb + r;
        if (m < NT)
          Obuf[((size_t)w_idx * NT + m) * Cc + head * HD + d] = (bf16_t)acc[j][r];
      }
    }
  }
}

// ---------------------------------------------------------------------------
// Host launcher
// ---------------------------------------------------------------------------
extern "C" void kernel_launch(void* const* d_in, const int* in_sizes, int n_in,
                              void* d_out, int out_size, void* d_ws, size_t ws_size,
                              hipStream_t stream) {
  const float* x      = (const float*)d_in[0];
  const float* ln1_g  = (const float*)d_in[1];
  const float* ln1_b  = (const float*)d_in[2];
  const float* qkv_w  = (const float*)d_in[3];
  const float* qkv_b  = (const float*)d_in[4];
  const float* proj_w = (const float*)d_in[5];
  const float* proj_b = (const float*)d_in[6];
  const float* rel_h  = (const float*)d_in[7];
  const float* rel_w  = (const float*)d_in[8];
  const float* ln2_g  = (const float*)d_in[9];
  const float* ln2_b  = (const float*)d_in[10];
  const float* fc1_w  = (const float*)d_in[11];
  const float* fc1_b  = (const float*)d_in[12];
  const float* fc2_w  = (const float*)d_in[13];
  const float* fc2_b  = (const float*)d_in[14];
  float* out = (float*)d_out;

  // carve workspace
  char* p = (char*)d_ws;
  auto carve = [&](size_t bytes) -> char* {
    char* r = p;
    p += (bytes + 255) & ~(size_t)255;
    return r;
  };
  bf16_t* qkv_wt = (bf16_t*)carve((size_t)Cc * 3 * Cc * 2);
  bf16_t* proj_wt= (bf16_t*)carve((size_t)Cc * Cc * 2);
  bf16_t* fc1_wt = (bf16_t*)carve((size_t)Cc * MLP * 2);
  bf16_t* fc2_wt = (bf16_t*)carve((size_t)MLP * Cc * 2);
  bf16_t* xw     = (bf16_t*)carve((size_t)ROWS * Cc * 2);
  bf16_t* qbuf   = (bf16_t*)carve((size_t)BW * NH * NT * HD * 2);
  bf16_t* kbuf   = (bf16_t*)carve((size_t)BW * NH * NT * HD * 2);
  bf16_t* vbuf   = (bf16_t*)carve((size_t)BW * NH * NT * HD * 2);
  bf16_t* obuf   = (bf16_t*)carve((size_t)ROWS * Cc * 2);
  float*  x1     = (float*)carve((size_t)TOK * Cc * 4);
  bf16_t* mbuf   = (bf16_t*)carve((size_t)TOK * Cc * 2);
  bf16_t* hbuf   = (bf16_t*)carve((size_t)TOK * MLP * 2);

  auto tgrid = [](size_t n) { return dim3((unsigned)((n + 255) / 256)); };

  // 1) weight transposes to bf16
  transpose_bf16_kernel<<<tgrid((size_t)Cc * 3 * Cc), 256, 0, stream>>>(qkv_w,  qkv_wt,  Cc,  3 * Cc);
  transpose_bf16_kernel<<<tgrid((size_t)Cc * Cc),     256, 0, stream>>>(proj_w, proj_wt, Cc,  Cc);
  transpose_bf16_kernel<<<tgrid((size_t)Cc * MLP),    256, 0, stream>>>(fc1_w,  fc1_wt,  Cc,  MLP);
  transpose_bf16_kernel<<<tgrid((size_t)MLP * Cc),    256, 0, stream>>>(fc2_w,  fc2_wt,  MLP, Cc);

  // 2) LN1 + window partition
  ln1_partition_kernel<<<dim3(ROWS), 256, 0, stream>>>(x, ln1_g, ln1_b, xw);

  // 3) QKV projection (WMMA + async B staging)
  gemm_bf16_kernel<0><<<dim3((ROWS + 127) / 128, (3 * Cc) / 64), 256, 0, stream>>>(
      xw, qkv_wt, qkv_b, ROWS, 3 * Cc, Cc, nullptr, nullptr, nullptr, qbuf, kbuf, vbuf);

  // 4) windowed attention with rel-pos bias (WMMA, 290KB LDS per workgroup)
  (void)hipFuncSetAttribute((const void*)attn_kernel,
                            hipFuncAttributeMaxDynamicSharedMemorySize, ATT_SMEM);
  attn_kernel<<<dim3(BW * NH), 256, ATT_SMEM, stream>>>(qbuf, kbuf, vbuf, rel_h, rel_w, obuf);

  // 5) output projection + unpartition + residual -> x1
  gemm_bf16_kernel<1><<<dim3((ROWS + 127) / 128, Cc / 64), 256, 0, stream>>>(
      obuf, proj_wt, proj_b, ROWS, Cc, Cc, x1, nullptr, x, nullptr, nullptr, nullptr);

  // 6) LN2
  ln2_kernel<<<dim3(TOK), 256, 0, stream>>>(x1, ln2_g, ln2_b, mbuf);

  // 7) fc1 + exact GELU
  gemm_bf16_kernel<2><<<dim3(TOK / 128, MLP / 64), 256, 0, stream>>>(
      mbuf, fc1_wt, fc1_b, TOK, MLP, Cc, nullptr, hbuf, nullptr, nullptr, nullptr, nullptr);

  // 8) fc2 + residual -> d_out
  gemm_bf16_kernel<3><<<dim3(TOK / 128, Cc / 64), 256, 0, stream>>>(
      hbuf, fc2_wt, fc2_b, TOK, Cc, MLP, out, nullptr, x1, nullptr, nullptr, nullptr);
}
